// Net_58411555226119
// MI455X (gfx1250) — matrix operand order repaired and checked by reading, not compiled, not run on hardware
//
#include <hip/hip_runtime.h>
#include <hip/hip_bf16.h>
#include <math.h>

typedef __attribute__((ext_vector_type(16))) _Float16 v16h;
typedef __attribute__((ext_vector_type(8)))  _Float16 v8h;
typedef __attribute__((ext_vector_type(8)))  float    v8f;

#define GB 32
#define GA 150
#define GS 1000
#define GD 200
#define GNB 6

__device__ __forceinline__ float lrelu_f(float x, float s) { return x >= 0.f ? x : s * x; }

// =====================================================================
// WMMA GEMM:  C[M,N] = act( op(A)[M,K] * op(B)[K,N] + bias ) * cmul
//   AT: A stored K x M.  BT: B stored N x K.  IDX: row-gather on A.
// Double-buffered LDS + register staging: global loads for tile i+1 are in
// flight while tile i's WMMAs execute; one barrier per k-step.
// Full k-steps have loop-invariant staging conditions (unswitchable);
// the single partial tail step uses branchless clamped loads.
// Block = 128 threads (4 waves); tile 64x64, K-step 32.
// NOTE: lda/ldb multiples of 4 (callers guarantee; u2in padded to 208).
// =====================================================================
template<bool AT, bool BT, bool LRELU, bool IDX, bool BIAS, bool CMUL>
__global__ __launch_bounds__(128)
void gemm_wmma(const float* __restrict__ Asrc, const float* __restrict__ Bsrc,
               const float* __restrict__ bias, const float* __restrict__ cmul,
               float* __restrict__ C, int M, int N, int K,
               int lda, int ldb, int ldc,
               long long bsA, long long bsB, long long bsC, long long bsMul,
               const int* __restrict__ a_idx, float slope)
{
    __shared__ __align__(16) _Float16 As[2][64][32];   // M x K tiles
    __shared__ __align__(16) _Float16 Bs[2][64][32];   // N x K tiles

    const int bz = blockIdx.z;
    Asrc += bsA * bz;
    Bsrc += bsB * bz;
    C    += bsC * bz;
    if constexpr (CMUL) cmul += bsMul * bz;

    const int tid  = threadIdx.x;
    const int wave = tid >> 5;
    const int lane = tid & 31;
    const int m0   = blockIdx.y * 64;
    const int n0   = blockIdx.x * 64;

    v8f acc[4] = {};

    // ---- staging thread coordinates ----
    const int ar  = tid >> 1, akc = (tid & 1) << 4;   // A normal
    const int akr = tid >> 2, amc = (tid & 3) << 4;   // A transposed
    const int bkr = tid >> 2, bnc = (tid & 3) << 4;   // B normal
    const int bnr = tid >> 1, bkc = (tid & 1) << 4;   // B transposed

    long long arow = 0;
    bool amv = false;
    if constexpr (!AT) {
        int m = m0 + ar;
        amv = (m < M);
        if (amv) arow = IDX ? (long long)a_idx[m] : (long long)m;
    }
    bool bnv = false;
    if constexpr (BT) bnv = (n0 + bnr < N);

    float ra[16], rb[16];

    // ---- register-staging loaders; `full` is a literal at each call site ----
    auto loadA = [&](int k0, bool full) {
        if constexpr (!AT) {
            const float* rowp = Asrc + arow * lda;       // arow==0 if !amv
            if (amv && full) {
                const float4* s4 = (const float4*)(rowp + k0 + akc);
                float4 f0 = s4[0], f1 = s4[1], f2 = s4[2], f3 = s4[3];
                ra[0]=f0.x; ra[1]=f0.y; ra[2]=f0.z; ra[3]=f0.w;
                ra[4]=f1.x; ra[5]=f1.y; ra[6]=f1.z; ra[7]=f1.w;
                ra[8]=f2.x; ra[9]=f2.y; ra[10]=f2.z; ra[11]=f2.w;
                ra[12]=f3.x; ra[13]=f3.y; ra[14]=f3.z; ra[15]=f3.w;
            } else {
                const int jmax = amv ? (K - k0 - akc) : 0;
                #pragma unroll
                for (int j = 0; j < 16; ++j) {
                    bool ok = (j < jmax);
                    int kk = ok ? (k0 + akc + j) : 0;
                    float v = rowp[kk];
                    ra[j] = ok ? v : 0.f;
                }
            }
        } else {
            const int  k  = k0 + akr;
            const bool kv = full || (k < K);
            const long long krow = (long long)(kv ? k : 0) * lda;
            const int jmax = kv ? (M - m0 - amc) : 0;
            if (kv && jmax >= 16) {
                const float4* s4 = (const float4*)(Asrc + krow + (m0 + amc));
                float4 f0 = s4[0], f1 = s4[1], f2 = s4[2], f3 = s4[3];
                ra[0]=f0.x; ra[1]=f0.y; ra[2]=f0.z; ra[3]=f0.w;
                ra[4]=f1.x; ra[5]=f1.y; ra[6]=f1.z; ra[7]=f1.w;
                ra[8]=f2.x; ra[9]=f2.y; ra[10]=f2.z; ra[11]=f2.w;
                ra[12]=f3.x; ra[13]=f3.y; ra[14]=f3.z; ra[15]=f3.w;
            } else {
                #pragma unroll
                for (int j = 0; j < 16; ++j) {
                    bool ok = (j < jmax);
                    int mm = ok ? (m0 + amc + j) : 0;
                    float v = Asrc[krow + mm];
                    ra[j] = ok ? v : 0.f;
                }
            }
        }
    };
    auto loadB = [&](int k0, bool full) {
        if constexpr (BT) {
            const float* rowp = Bsrc + (long long)(bnv ? (n0 + bnr) : 0) * ldb;
            if (bnv && full) {
                const float4* s4 = (const float4*)(rowp + k0 + bkc);
                float4 f0 = s4[0], f1 = s4[1], f2 = s4[2], f3 = s4[3];
                rb[0]=f0.x; rb[1]=f0.y; rb[2]=f0.z; rb[3]=f0.w;
                rb[4]=f1.x; rb[5]=f1.y; rb[6]=f1.z; rb[7]=f1.w;
                rb[8]=f2.x; rb[9]=f2.y; rb[10]=f2.z; rb[11]=f2.w;
                rb[12]=f3.x; rb[13]=f3.y; rb[14]=f3.z; rb[15]=f3.w;
            } else {
                const int jmax = bnv ? (K - k0 - bkc) : 0;
                #pragma unroll
                for (int j = 0; j < 16; ++j) {
                    bool ok = (j < jmax);
                    int kk = ok ? (k0 + bkc + j) : 0;
                    float v = rowp[kk];
                    rb[j] = ok ? v : 0.f;
                }
            }
        } else {
            const int  k  = k0 + bkr;
            const bool kv = full || (k < K);
            const long long krow = (long long)(kv ? k : 0) * ldb;
            const int jmax = kv ? (N - n0 - bnc) : 0;
            if (kv && jmax >= 16) {
                const float4* s4 = (const float4*)(Bsrc + krow + (n0 + bnc));
                float4 f0 = s4[0], f1 = s4[1], f2 = s4[2], f3 = s4[3];
                rb[0]=f0.x; rb[1]=f0.y; rb[2]=f0.z; rb[3]=f0.w;
                rb[4]=f1.x; rb[5]=f1.y; rb[6]=f1.z; rb[7]=f1.w;
                rb[8]=f2.x; rb[9]=f2.y; rb[10]=f2.z; rb[11]=f2.w;
                rb[12]=f3.x; rb[13]=f3.y; rb[14]=f3.z; rb[15]=f3.w;
            } else {
                #pragma unroll
                for (int j = 0; j < 16; ++j) {
                    bool ok = (j < jmax);
                    int nn = ok ? (n0 + bnc + j) : 0;
                    float v = Bsrc[krow + nn];
                    rb[j] = ok ? v : 0.f;
                }
            }
        }
    };
    auto storeA = [&](int p) {
        if constexpr (!AT) {
            v8h x, y;
            #pragma unroll
            for (int j = 0; j < 8; ++j) { x[j] = (_Float16)ra[j]; y[j] = (_Float16)ra[8 + j]; }
            v8h* d = (v8h*)&As[p][ar][akc];
            d[0] = x; d[1] = y;
        } else {
            #pragma unroll
            for (int j = 0; j < 16; ++j) As[p][amc + j][akr] = (_Float16)ra[j];
        }
    };
    auto storeB = [&](int p) {
        if constexpr (BT) {
            v8h x, y;
            #pragma unroll
            for (int j = 0; j < 8; ++j) { x[j] = (_Float16)rb[j]; y[j] = (_Float16)rb[8 + j]; }
            v8h* d = (v8h*)&Bs[p][bnr][bkc];
            d[0] = x; d[1] = y;
        } else {
            #pragma unroll
            for (int j = 0; j < 16; ++j) Bs[p][bnc + j][bkr] = (_Float16)rb[j];
        }
    };

    // ---- pipelined main loop ----
    const int nit = (K + 31) >> 5;
    if (32 <= K) { loadA(0, true);  loadB(0, true);  }
    else         { loadA(0, false); loadB(0, false); }

    int p = 0;
    for (int it = 0; it < nit; ++it) {
        storeA(p);
        storeB(p);
        __syncthreads();

        const int k1 = (it + 1) << 5;
        if (it + 1 < nit) {                       // prefetch next tile into regs
            if (k1 + 32 <= K) { loadA(k1, true);  loadB(k1, true);  }
            else              { loadA(k1, false); loadB(k1, false); }
        }

        // ---- WMMA: hoist all fragment loads, then 4 WMMAs ----
        const int fm = lane & 15;
        const int kh = (lane >> 4) << 3;   // A: K-half offset 0/8
        const int kb = (lane >> 4) << 3;   // B: K-half offset 0/8
        v8h alo = *(const v8h*)&As[p][(wave << 4) + fm][kh];
        v8h ahi = *(const v8h*)&As[p][(wave << 4) + fm][16 + kh];
        v8h bl[4], bh[4];
        #pragma unroll
        for (int t = 0; t < 4; ++t) {
            bl[t] = *(const v8h*)&Bs[p][(t << 4) + fm][kb];
            bh[t] = *(const v8h*)&Bs[p][(t << 4) + fm][16 + kb];
        }
        v16h af = __builtin_shufflevector(alo, ahi,
                    0,1,2,3,4,5,6,7,8,9,10,11,12,13,14,15);
        #pragma unroll
        for (int t = 0; t < 4; ++t) {
            v16h bf = __builtin_shufflevector(bl[t], bh[t],
                        0,1,2,3,4,5,6,7,8,9,10,11,12,13,14,15);
            acc[t] = __builtin_amdgcn_wmma_f32_16x16x32_f16(
                         false, af, false, bf, (short)0, acc[t], false, false);
        }
        p ^= 1;
    }

    // ---------------- epilogue ----------------
    const int fm = lane & 15;
    const int mh = (lane >> 4) << 3;
    #pragma unroll
    for (int t = 0; t < 4; ++t) {
        int n = n0 + (t << 4) + fm;
        if (n < N) {
            #pragma unroll
            for (int r = 0; r < 8; ++r) {
                int m = m0 + (wave << 4) + mh + r;
                if (m < M) {
                    float v = acc[t][r];
                    if constexpr (BIAS)  v += bias[n];
                    if constexpr (LRELU) v = lrelu_f(v, slope);
                    long long ci = (long long)m * ldc + n;
                    if constexpr (CMUL)  v *= cmul[ci];
                    C[ci] = v;
                }
            }
        }
    }
}

// =====================================================================
// Small helper kernels
// =====================================================================
__global__ void gather_ei_kernel(const int* __restrict__ fb,
                                 const float* __restrict__ emb_bond,
                                 float* __restrict__ ei, int n)
{
    int i = blockIdx.x * blockDim.x + threadIdx.x;
    if (i >= n) return;                       // n = B*A*NB
    int row = i / GNB, j = i % GNB;
    ei[i] = emb_bond[fb[row] * GNB + j];
}

// u2in padded to 208 columns (multiple of 4 for vector staging).
__global__ void build_u2in_kernel(const float* __restrict__ vf, const float* __restrict__ ei,
                                  const int* __restrict__ anb, const int* __restrict__ bnb,
                                  float* __restrict__ u2in, int n)
{
    int i = blockIdx.x * blockDim.x + threadIdx.x;
    if (i >= n) return;                       // n = B*A*NB*208
    int row = i / 208, c = i % 208;
    float v = 0.f;
    if (c < GD)            v = vf[(long long)anb[row] * GD + c];
    else if (c < GD + GNB) v = ei[bnb[row] * GNB + (c - GD)];
    u2in[i] = v;
}

__global__ void reduce_nl_kernel(const float* __restrict__ nlfull,
                                 const float* __restrict__ nbs,
                                 float* __restrict__ nl, int n)
{
    int i = blockIdx.x * blockDim.x + threadIdx.x;
    if (i >= n) return;                       // n = B*A*D
    int p = i / GD, d = i % GD;
    float s = 0.f;
    #pragma unroll
    for (int j = 0; j < GNB; ++j)
        s += nlfull[(long long)(p * GNB + j) * GD + d] * nbs[p * GNB + j];
    nl[i] = s;
}

__global__ void build_u1in_kernel(const float* __restrict__ vf, const float* __restrict__ nl,
                                  float* __restrict__ u1in, int n)
{
    int i = blockIdx.x * blockDim.x + threadIdx.x;
    if (i >= n) return;                       // n = B*A*2D
    int p = i / (2 * GD), c = i % (2 * GD);
    u1in[i] = (c < GD) ? vf[(long long)p * GD + c] : nl[(long long)p * GD + (c - GD)];
}

// Fused: conv2 & conv3 (1->16ch, 3x3 pad1) + BN + lrelu(0.01) + 16->1 pointwise
// + mask; writes sigmoid(pp_fin) into d_out and mulMat to workspace.
__global__ void ppconv_kernel(const float* __restrict__ pp,
    const float* __restrict__ Lmask, const float* __restrict__ Pmask,
    const float* __restrict__ w2, const float* __restrict__ g2, const float* __restrict__ bb2,
    const float* __restrict__ w3, const float* __restrict__ g3, const float* __restrict__ bb3,
    const float* __restrict__ Wpw, const float* __restrict__ bpw,
    const float* __restrict__ Wmm, const float* __restrict__ bmm,
    float* __restrict__ outSig, float* __restrict__ mulMat)
{
    int s = blockIdx.x * blockDim.x + threadIdx.x;
    int a = blockIdx.y, b = blockIdx.z;
    if (s >= GS) return;
    float x[9];
    #pragma unroll
    for (int dy = 0; dy < 3; ++dy)
        #pragma unroll
        for (int dx = 0; dx < 3; ++dx) {
            int aa = a + dy - 1, ss = s + dx - 1;
            x[dy * 3 + dx] = (aa >= 0 && aa < GA && ss >= 0 && ss < GS)
                             ? pp[((long long)b * GA + aa) * GS + ss] : 0.f;
        }
    const float inv = rsqrtf(1.f + 1e-5f);
    float acc2 = 0.f, acc3 = 0.f;
    #pragma unroll
    for (int c = 0; c < 16; ++c) {
        float c2 = 0.f, c3 = 0.f;
        #pragma unroll
        for (int t = 0; t < 9; ++t) {
            c2 += w2[c * 9 + t] * x[t];
            c3 += w3[c * 9 + t] * x[t];
        }
        acc2 += lrelu_f(c2 * (g2[c] * inv) + bb2[c], 0.01f) * Wpw[c];
        acc3 += lrelu_f(c3 * (g3[c] * inv) + bb3[c], 0.01f) * Wmm[c];
    }
    float pm = Lmask[b * GA + a] * Pmask[b * GS + s];
    long long o = ((long long)b * GA + a) * GS + s;
    float v = (acc2 + bpw[0]) * pm;
    outSig[o] = 1.f / (1.f + __expf(-v));
    mulMat[o] = (acc3 + bmm[0]) * pm;
}

// Direct 3x3 conv, pad 1, NCHW. grid = (ceil(H*W/256), Cout, B)
__global__ void conv3x3_kernel(const float* __restrict__ in, const float* __restrict__ w,
                               float* __restrict__ out, int Cin, int H, int W)
{
    int pix = blockIdx.x * blockDim.x + threadIdx.x;
    int co = blockIdx.y, b = blockIdx.z, Cout = gridDim.y;
    if (pix >= H * W) return;
    int y = pix / W, x = pix % W;
    float acc = 0.f;
    for (int ci = 0; ci < Cin; ++ci) {
        const float* ip = in + (long long)(b * Cin + ci) * H * W;
        const float* wp = w + (long long)(co * Cin + ci) * 9;
        #pragma unroll
        for (int dy = 0; dy < 3; ++dy) {
            int yy = y + dy - 1;
            if (yy < 0 || yy >= H) continue;
            #pragma unroll
            for (int dx = 0; dx < 3; ++dx) {
                int xx = x + dx - 1;
                if (xx < 0 || xx >= W) continue;
                acc += wp[dy * 3 + dx] * ip[yy * W + xx];
            }
        }
    }
    out[(long long)(b * Cout + co) * H * W + pix] = acc;
}

// Maxpool 3x3 stride2 pad1 + BN + lrelu(0.01); optional NHWC output (free flatten).
__global__ void pool_bn_lrelu_kernel(const float* __restrict__ in,
    const float* __restrict__ g, const float* __restrict__ bt,
    float* __restrict__ out, int Hin, int Win, int Hout, int Wout, int nhwc)
{
    int pix = blockIdx.x * blockDim.x + threadIdx.x;
    int c = blockIdx.y, b = blockIdx.z, C = gridDim.y;
    if (pix >= Hout * Wout) return;
    int y = pix / Wout, x = pix % Wout;
    const float* ip = in + (long long)(b * C + c) * Hin * Win;
    float m = -INFINITY;
    #pragma unroll
    for (int dy = 0; dy < 3; ++dy) {
        int yy = 2 * y + dy - 1;
        if (yy < 0 || yy >= Hin) continue;
        #pragma unroll
        for (int dx = 0; dx < 3; ++dx) {
            int xx = 2 * x + dx - 1;
            if (xx < 0 || xx >= Win) continue;
            float v = ip[yy * Win + xx];
            m = v > m ? v : m;
        }
    }
    float v = lrelu_f(m * (g[c] * rsqrtf(1.f + 1e-5f)) + bt[c], 0.01f);
    long long o = nhwc ? (((long long)b * Hout + y) * Wout + x) * C + c
                       : ((long long)(b * C + c) * Hout + y) * Wout + x;
    out[o] = v;
}

// Final tiny FC tail: fin[b] = ((fc1 @ Wf1 + bf1) @ Wf2 + bf2)
__global__ void final_fc_kernel(const float* __restrict__ fc1, const float* __restrict__ Wf1,
    const float* __restrict__ bf1, const float* __restrict__ Wf2, const float* __restrict__ bf2,
    float* __restrict__ fin)
{
    int b = threadIdx.x;
    if (b >= GB) return;
    float r = bf2[0];
    for (int j = 0; j < 32; ++j) {
        float t = bf1[j];
        for (int k = 0; k < 128; ++k) t += fc1[b * 128 + k] * Wf1[k * 32 + j];
        r += t * Wf2[j];
    }
    fin[b] = r;
}

// =====================================================================
// Host launcher
// =====================================================================
extern "C" void kernel_launch(void* const* d_in, const int* in_sizes, int n_in,
                              void* d_out, int out_size, void* d_ws, size_t ws_size,
                              hipStream_t stream)
{
    const float* L_mask   = (const float*)d_in[0];
    const int*   vecL     = (const int*)  d_in[1];
    const int*   fb       = (const int*)  d_in[2];
    const int*   anb      = (const int*)  d_in[3];
    const int*   bnb      = (const int*)  d_in[4];
    const float* nbs      = (const float*)d_in[5];
    const int*   vecP     = (const int*)  d_in[6];
    const float* P_mask   = (const float*)d_in[7];
    const float* tmpFRI   = (const float*)d_in[8];
    const float* emb_atom = (const float*)d_in[9];
    const float* emb_bond = (const float*)d_in[10];
    const float* W_fc1p = (const float*)d_in[11];
    const float* b_fc1p = (const float*)d_in[12];
    const float* W_fc1l = (const float*)d_in[13];
    const float* b_fc1l = (const float*)d_in[14];
    const float* W_u2   = (const float*)d_in[15];
    const float* b_u2   = (const float*)d_in[16];
    const float* W_u1   = (const float*)d_in[17];
    const float* b_u1   = (const float*)d_in[18];
    const float* conv2_w = (const float*)d_in[19];
    const float* conv2_g = (const float*)d_in[20];
    const float* conv2_b = (const float*)d_in[21];
    const float* conv3_w = (const float*)d_in[22];
    const float* conv3_g = (const float*)d_in[23];
    const float* conv3_b = (const float*)d_in[24];
    const float* Wpip_pw = (const float*)d_in[25];
    const float* bpip_pw = (const float*)d_in[26];
    const float* Wpip_mm = (const float*)d_in[27];
    const float* bpip_mm = (const float*)d_in[28];
    const float* c1_w = (const float*)d_in[29];
    const float* c1_g = (const float*)d_in[30];
    const float* c1_b = (const float*)d_in[31];
    const float* c2_w = (const float*)d_in[32];
    const float* c2_g = (const float*)d_in[33];
    const float* c2_b = (const float*)d_in[34];
    const float* c3_w = (const float*)d_in[35];
    const float* c3_g = (const float*)d_in[36];
    const float* c3_b = (const float*)d_in[37];
    const float* W_cnn1 = (const float*)d_in[38];
    const float* b_cnn1 = (const float*)d_in[39];
    const float* W_f1 = (const float*)d_in[40];
    const float* b_f1 = (const float*)d_in[41];
    const float* W_f2 = (const float*)d_in[42];
    const float* b_f2 = (const float*)d_in[43];

    float* out = (float*)d_out;           // [0,32): fin ; [32,...): sigmoid(pp_fin)
    float* ws  = (float*)d_ws;

    // ---- workspace layout (floats); CNN phase overlays dead GNN buffers ----
    const long long OFF_VF   = 0;
    const long long OFF_EI   = OFF_VF   +  960000LL;   // B*A*D
    const long long OFF_U2IN = OFF_EI   +   28800LL;   // B*A*NB
    const long long OFF_NLF  = OFF_U2IN + 5990400LL;   // B*A*NB*208 (padded)
    const long long OFF_NL   = OFF_NLF  + 5760000LL;   // B*A*NB*D
    const long long OFF_U1IN = OFF_NL   +  960000LL;
    const long long OFF_GF   = OFF_U1IN + 1920000LL;
    const long long OFF_PFC  = OFF_GF   +  960000LL;   // B*S*D
    const long long OFF_PP   = OFF_PFC  + 6400000LL;   // B*A*S
    const long long OFF_MM   = OFF_PP   + 4800000LL;   // B*A*S
    const long long OFF_LTOP = OFF_MM   + 4800000LL;   // B*D*S
    const long long OFF_LPM  = OFF_LTOP + 6400000LL;   // B*D*D
    const long long OFF_C1   = 0;                       // reuse front
    const long long OFF_P1   = OFF_C1  + 20480000LL;   // 32*16*200*200
    const long long OFF_C2   = OFF_P1  +  5120000LL;   // 32*16*100*100
    const long long OFF_P2   = OFF_C2  + 10240000LL;   // ends < OFF_LPM
    const long long OFF_C3   = OFF_LPM +  1280000LL;   // after LPMul
    const long long OFF_P3   = OFF_C3  +  5120000LL;   // 32*64*50*50
    const long long OFF_FC1  = OFF_P3  +  1280000LL;   // 32*64*25*25 flat NHWC

    float* vf     = ws + OFF_VF;
    float* ei     = ws + OFF_EI;
    float* u2in   = ws + OFF_U2IN;
    float* nlfull = ws + OFF_NLF;
    float* nl     = ws + OFF_NL;
    float* u1in   = ws + OFF_U1IN;
    float* gf     = ws + OFF_GF;
    float* PFC    = ws + OFF_PFC;
    float* pp     = ws + OFF_PP;
    float* mulMat = ws + OFF_MM;
    float* LtoP   = ws + OFF_LTOP;
    float* LPMul  = ws + OFF_LPM;
    float* c1o    = ws + OFF_C1;
    float* p1     = ws + OFF_P1;
    float* c2o    = ws + OFF_C2;
    float* p2     = ws + OFF_P2;
    float* c3o    = ws + OFF_C3;
    float* p3flat = ws + OFF_P3;
    float* fc1    = ws + OFF_FC1;

    // ---- GNN front ----
    gather_ei_kernel<<<dim3((28800 + 255) / 256), 256, 0, stream>>>(fb, emb_bond, ei, 28800);

    // vf = lrelu(emb_atom[vecL] @ W_fc1l + b): M=4800 K=200 N=200, gather fused
    gemm_wmma<false, false, true, true, true, false>
        <<<dim3(4, 75, 1), 128, 0, stream>>>(
            emb_atom, W_fc1l, b_fc1l, nullptr, vf,
            4800, GD, GD, GD, GD, GD, 0, 0, 0, 0, vecL, 0.1f);

    build_u2in_kernel<<<dim3((5990400 + 255) / 256), 256, 0, stream>>>(vf, ei, anb, bnb, u2in, 5990400);

    // nlfull = lrelu(u2in @ W_u2 + b): M=28800 K=206 N=200 (lda padded to 208)
    gemm_wmma<false, false, true, false, true, false>
        <<<dim3(4, 450, 1), 128, 0, stream>>>(
            u2in, W_u2, b_u2, nullptr, nlfull,
            28800, GD, GD + GNB, 208, GD, GD, 0, 0, 0, 0, nullptr, 0.1f);

    reduce_nl_kernel<<<dim3((960000 + 255) / 256), 256, 0, stream>>>(nlfull, nbs, nl, 960000);
    build_u1in_kernel<<<dim3((1920000 + 255) / 256), 256, 0, stream>>>(vf, nl, u1in, 1920000);

    // gf = lrelu(u1in @ W_u1 + b): M=4800 K=400 N=200
    gemm_wmma<false, false, true, false, true, false>
        <<<dim3(4, 75, 1), 128, 0, stream>>>(
            u1in, W_u1, b_u1, nullptr, gf,
            4800, GD, 2 * GD, 2 * GD, GD, GD, 0, 0, 0, 0, nullptr, 0.1f);

    // PFC = emb_atom[vecP] @ W_fc1p + b: M=32000 K=200 N=200, gather fused, linear
    gemm_wmma<false, false, false, true, true, false>
        <<<dim3(4, 500, 1), 128, 0, stream>>>(
            emb_atom, W_fc1p, b_fc1p, nullptr, PFC,
            GB * GS, GD, GD, GD, GD, GD, 0, 0, 0, 0, vecP, 0.f);

    // pp[b] = (vf[b] @ PFC[b]^T) * tmpFRI[b]: M=150 N=1000 K=200, B stored NxK
    gemm_wmma<false, true, false, false, false, true>
        <<<dim3(16, 3, GB), 128, 0, stream>>>(
            vf, PFC, nullptr, tmpFRI, pp,
            GA, GS, GD, GD, GD, GS,
            (long long)GA * GD, (long long)GS * GD, (long long)GA * GS, (long long)GA * GS,
            nullptr, 0.f);

    // fused conv2/conv3 + BN + lrelu + pointwise + mask (+ sigmoid into d_out)
    ppconv_kernel<<<dim3((GS + 255) / 256, GA, GB), 256, 0, stream>>>(
        pp, L_mask, P_mask, conv2_w, conv2_g, conv2_b, conv3_w, conv3_g, conv3_b,
        Wpip_pw, bpip_pw, Wpip_mm, bpip_mm, out + 32, mulMat);

    // LtoP[b] = gf[b]^T @ mulMat[b]: M=200 N=1000 K=150, A stored KxM
    gemm_wmma<true, false, false, false, false, false>
        <<<dim3(16, 4, GB), 128, 0, stream>>>(
            gf, mulMat, nullptr, nullptr, LtoP,
            GD, GS, GA, GD, GS, GS,
            (long long)GA * GD, (long long)GA * GS, (long long)GD * GS, 0,
            nullptr, 0.f);

    // LPMul[b] = LtoP[b] @ PFC[b]: M=200 N=200 K=1000
    gemm_wmma<false, false, false, false, false, false>
        <<<dim3(4, 4, GB), 128, 0, stream>>>(
            LtoP, PFC, nullptr, nullptr, LPMul,
            GD, GD, GS, GS, GD, GD,
            (long long)GD * GS, (long long)GS * GD, (long long)GD * GD, 0,
            nullptr, 0.f);

    // ---- CNN tail ----
    conv3x3_kernel<<<dim3((200 * 200 + 255) / 256, 16, GB), 256, 0, stream>>>(LPMul, c1_w, c1o, 1, 200, 200);
    pool_bn_lrelu_kernel<<<dim3((100 * 100 + 255) / 256, 16, GB), 256, 0, stream>>>(c1o, c1_g, c1_b, p1, 200, 200, 100, 100, 0);
    conv3x3_kernel<<<dim3((100 * 100 + 255) / 256, 32, GB), 256, 0, stream>>>(p1, c2_w, c2o, 16, 100, 100);
    pool_bn_lrelu_kernel<<<dim3((50 * 50 + 255) / 256, 32, GB), 256, 0, stream>>>(c2o, c2_g, c2_b, p2, 100, 100, 50, 50, 0);
    conv3x3_kernel<<<dim3((50 * 50 + 255) / 256, 64, GB), 256, 0, stream>>>(p2, c3_w, c3o, 32, 50, 50);
    pool_bn_lrelu_kernel<<<dim3((25 * 25 + 255) / 256, 64, GB), 256, 0, stream>>>(c3o, c3_g, c3_b, p3flat, 50, 50, 25, 25, 1);

    // fc1 = p3flat(32x40000) @ W_cnn1(40000x128) + b_cnn1
    gemm_wmma<false, false, false, false, true, false>
        <<<dim3(2, 1, 1), 128, 0, stream>>>(
            p3flat, W_cnn1, b_cnn1, nullptr, fc1,
            GB, 128, 40000, 40000, 128, 128, 0, 0, 0, 0, nullptr, 0.f);

    // fin = (fc1 @ W_f1 + b_f1) @ W_f2 + b_f2
    final_fc_kernel<<<1, 32, 0, stream>>>(fc1, W_f1, b_f1, W_f2, b_f2, out);
}
